// MMWindowAttention_10110353015206
// MI455X (gfx1250) — compile-verified, matrix operand order checked by reading, not run on hardware
//
#include <hip/hip_runtime.h>

// ---------------------------------------------------------------------------
// MMWindowAttention for MI455X (gfx1250, wave32, WMMA bf16 path)
//  - all matmuls (QKV, Q.K^T, P.V, out-proj) on v_wmma_f32_16x16x32_bf16
//  - LDS staging via global_load_async_to_lds_b128 (ASYNCcnt) where no
//    transpose is needed; double-buffered, software-pipelined GEMM K loop
// ---------------------------------------------------------------------------
#define S_TOK   18432          // T*H*W
#define L_TXT   256
#define N_HEADS 16
#define HD      96
#define DIM     1536
#define QKV_N   4608
#define NWIN    72             // nt*nh*nw = 2*6*6
#define WIN     256            // TT*HH*WW = 4*8*8
#define ATTN_SCALE 0.10206207261596575f   // 1/sqrt(96)

typedef __attribute__((ext_vector_type(16))) __bf16 v16bf;
typedef __attribute__((ext_vector_type(8)))  float  v8f;

union Frag16 { unsigned u[8]; v16bf v; };

__device__ __forceinline__ unsigned short f2bf(float f) {
  union { float f; unsigned u; } x; x.f = f;
  unsigned r = x.u + 0x7FFFu + ((x.u >> 16) & 1u);   // round-to-nearest-even
  return (unsigned short)(r >> 16);
}

// ISA 16-bit A/B fragment K index for element-pair p given lane K-group kg:
// VGPR 0..3 -> kg+{0,2,4,6}, VGPR 4..7 -> kg+16+{0,2,4,6} (pairs are K,K+1)
__device__ __forceinline__ int k0_of(int p, int kg) {
  return kg + ((p < 4) ? (2 * p) : (16 + 2 * (p - 4)));
}

// CDNA5 async global->LDS copy (16B per lane), tracked with ASYNCcnt.
// Generic shared pointer low 32 bits == LDS byte address (aperture in hi bits).
__device__ __forceinline__ void async_copy16(const unsigned short* lds_dst,
                                             const void* gsrc) {
  unsigned loff = (unsigned)(unsigned long long)lds_dst;
  asm volatile("global_load_async_to_lds_b128 %0, %1, off"
               :: "v"(loff), "v"(gsrc)
               : "memory");
}
__device__ __forceinline__ void async_wait0() {
  asm volatile("s_wait_asynccnt 0x0" ::: "memory");
}

// ---------------------------------------------------------------------------
// Elementwise helpers
// ---------------------------------------------------------------------------
__global__ void cast_bf16_kernel(const float* __restrict__ in,
                                 unsigned short* __restrict__ out, long n) {
  long i = (long)blockIdx.x * blockDim.x + threadIdx.x;
  long stride = (long)gridDim.x * blockDim.x;
  for (; i < n; i += stride) out[i] = f2bf(in[i]);
}

__global__ void maskbias_kernel(const unsigned char* __restrict__ mask,
                                float* __restrict__ out) {
  int i = blockIdx.x * blockDim.x + threadIdx.x;
  if (i < L_TXT) out[i] = mask[i] ? 0.f : -1e30f;
}

// ---------------------------------------------------------------------------
// Generic bf16 x bf16 -> f32 GEMM.  Block tile 128x128, 8 waves (4x2),
// wave tile 32x64 = 2x4 WMMA tiles, K staged 32 wide, double-buffered LDS.
// A staged with async-to-LDS; B staged transposed (regs -> b16 scatter) so
// every bf16 fragment pair is a contiguous dword LDS load.
// ---------------------------------------------------------------------------
__global__ __launch_bounds__(256) void gemm_bf16_kernel(
    const unsigned short* __restrict__ A,   // [M][K] bf16
    const unsigned short* __restrict__ Bw,  // [K][N] bf16
    float* __restrict__ C,                  // [M][N] f32
    const float* __restrict__ bias,         // [N] or nullptr
    int M, int N, int K)
{
  __shared__ __align__(16) unsigned short ldsA[2][128 * 32];   // [m][k]
  __shared__ __align__(16) unsigned short ldsBt[2][128 * 32];  // [n][k]

  const int tid   = threadIdx.x;
  const int lane  = tid & 31;
  const int wid   = tid >> 5;
  const int waveM = wid >> 1;     // 0..3
  const int waveN = wid & 1;      // 0..1
  const int ln15  = lane & 15;
  const int kg    = (lane >> 4) * 8;
  const size_t rowBase = (size_t)blockIdx.y * 128;
  const size_t colBase = (size_t)blockIdx.x * 128;

  v8f zero = {0.f, 0.f, 0.f, 0.f, 0.f, 0.f, 0.f, 0.f};
  v8f acc[2][4];
#pragma unroll
  for (int mt = 0; mt < 2; mt++)
#pragma unroll
    for (int nt = 0; nt < 4; nt++) acc[mt][nt] = zero;

  uint4 breg[2];

  auto issueA = [&](int buf, int kt) {            // async global -> LDS
#pragma unroll
    for (int c = tid; c < 512; c += 256) {
      int r = c >> 2, o = (c & 3) * 8;
      async_copy16(&ldsA[buf][r * 32 + o],
                   &A[(rowBase + r) * (size_t)K + kt + o]);
    }
  };
  auto loadB = [&](int kt) {                      // global -> regs
#pragma unroll
    for (int i = 0; i < 2; i++) {
      int c = tid + i * 256;
      int kk = c >> 4, nc = (c & 15) * 8;
      breg[i] = *(const uint4*)&Bw[(size_t)(kt + kk) * N + colBase + nc];
    }
  };
  auto scatterB = [&](int buf) {                  // regs -> LDS transposed
#pragma unroll
    for (int i = 0; i < 2; i++) {
      int c = tid + i * 256;
      int kk = c >> 4, nc = (c & 15) * 8;
      union { uint4 d4; unsigned short e[8]; } u; u.d4 = breg[i];
#pragma unroll
      for (int j = 0; j < 8; j++) ldsBt[buf][(nc + j) * 32 + kk] = u.e[j];
    }
  };
  auto compute = [&](int buf) {
    Frag16 af[2], bfr[4];
#pragma unroll
    for (int mt = 0; mt < 2; mt++) {
      int m = waveM * 32 + mt * 16 + ln15;
#pragma unroll
      for (int p = 0; p < 8; p++)
        af[mt].u[p] = *(const unsigned*)&ldsA[buf][m * 32 + k0_of(p, kg)];
    }
#pragma unroll
    for (int nt = 0; nt < 4; nt++) {
      int n = waveN * 64 + nt * 16 + ln15;
#pragma unroll
      for (int p = 0; p < 8; p++)
        bfr[nt].u[p] = *(const unsigned*)&ldsBt[buf][n * 32 + k0_of(p, kg)];
    }
#pragma unroll
    for (int mt = 0; mt < 2; mt++)
#pragma unroll
      for (int nt = 0; nt < 4; nt++)
        acc[mt][nt] = __builtin_amdgcn_wmma_f32_16x16x32_bf16(
            false, af[mt].v, false, bfr[nt].v, (short)0, acc[mt][nt], false, false);
  };

  // prologue: fill buffer 0
  issueA(0, 0);
  loadB(0);
  scatterB(0);
  async_wait0();
  __syncthreads();

  for (int kt = 0; kt < K; kt += 32) {
    const int cur = (kt >> 5) & 1, nxt = cur ^ 1;
    const bool more = (kt + 32) < K;
    if (more) {
      issueA(nxt, kt + 32);       // async, overlaps with compute below
      loadB(kt + 32);
      if (kt + 64 < K) {          // prefetch two tiles ahead into GL2
        __builtin_prefetch(&A[(rowBase + (tid >> 1)) * (size_t)K + kt + 64], 0, 1);
        __builtin_prefetch(&Bw[(size_t)(kt + 64 + (tid >> 4)) * N + colBase], 0, 1);
      }
    }
    compute(cur);
    if (more) scatterB(nxt);
    async_wait0();
    __syncthreads();
  }

  // epilogue: C layout per ISA — VGPR r: lanes 0-15 m=r, lanes 16-31 m=r+8
#pragma unroll
  for (int mt = 0; mt < 2; mt++)
#pragma unroll
    for (int nt = 0; nt < 4; nt++)
#pragma unroll
      for (int r = 0; r < 8; r++) {
        size_t gm = rowBase + waveM * 32 + mt * 16 + r + ((lane >> 4) << 3);
        size_t gn = colBase + waveN * 64 + nt * 16 + ln15;
        float v = acc[mt][nt][r];
        if (bias) v += bias[gn];
        C[gm * (size_t)N + gn] = v;
      }
}

// ---------------------------------------------------------------------------
// RMS + RoPE + window permutation (vid), RMS only (txt). f32 in, bf16 out.
// ---------------------------------------------------------------------------
__device__ __forceinline__ void rms96(float* x, const float* __restrict__ g) {
  float ss = 0.f;
#pragma unroll
  for (int d = 0; d < 96; d++) ss += x[d] * x[d];
  float r = rsqrtf(ss * (1.f / 96.f) + 1e-6f);
#pragma unroll
  for (int d = 0; d < 96; d++) x[d] *= r * g[d];
}

__device__ __forceinline__ void rope96(float* x, int t, int h, int w) {
  // inv[j] = 10000^(-j/16), constant-folded under full unroll
  const float invt[16] = {
    1.0f,           0.5623413252f,  0.3162277660f,  0.1778279410f,
    0.1f,           0.05623413252f, 0.03162277660f, 0.01778279410f,
    0.01f,          0.005623413252f,0.003162277660f,0.001778279410f,
    0.001f,         0.0005623413252f,0.0003162277660f,0.0001778279410f };
#pragma unroll
  for (int d = 0; d < 48; d++) {
    int grp = d >> 4, j = d & 15;
    float pos = (grp == 0) ? (float)t : (grp == 1) ? (float)h : (float)w;
    float sn, cs;
    __sincosf(pos * invt[j], &sn, &cs);
    float a = x[d], b = x[d + 48];
    x[d]      = a * cs - b * sn;   // x*cos + rot_half(x)*sin, first half
    x[d + 48] = b * cs + a * sn;   // second half
  }
}

__global__ __launch_bounds__(256) void post_vid_kernel(
    const float* __restrict__ qkv,                 // [S][3*DIM]
    const float* __restrict__ gq, const float* __restrict__ gk,
    unsigned short* __restrict__ qw, unsigned short* __restrict__ kw,
    unsigned short* __restrict__ vw)               // [head][win][row][96]
{
  int idx = blockIdx.x * 256 + threadIdx.x;
  if (idx >= S_TOK * N_HEADS) return;
  int s = idx >> 4, head = idx & 15;
  int t = s / (48 * 48), hs = (s / 48) % 48, wsp = s % 48;
  int it = t >> 2, tt = t & 3, ih = hs >> 3, hh = hs & 7, iw = wsp >> 3, ww = wsp & 7;
  int window = (it * 6 + ih) * 6 + iw;
  int rrow   = (tt * 8 + hh) * 8 + ww;
  size_t outOff = (((size_t)head * NWIN + window) * WIN + rrow) * HD;
  const float* base = qkv + (size_t)s * QKV_N + head * HD;

  float x[96];
#pragma unroll
  for (int d = 0; d < 96; d++) x[d] = base[d];
  rms96(x, gq); rope96(x, t, hs, wsp);
#pragma unroll
  for (int d = 0; d < 96; d++) qw[outOff + d] = f2bf(x[d]);

#pragma unroll
  for (int d = 0; d < 96; d++) x[d] = base[DIM + d];
  rms96(x, gk); rope96(x, t, hs, wsp);
#pragma unroll
  for (int d = 0; d < 96; d++) kw[outOff + d] = f2bf(x[d]);

#pragma unroll
  for (int d = 0; d < 96; d++) vw[outOff + d] = f2bf(base[2 * DIM + d]);
}

__global__ __launch_bounds__(256) void post_txt_kernel(
    const float* __restrict__ qkv,                 // [L][3*DIM]
    const float* __restrict__ gq, const float* __restrict__ gk,
    unsigned short* __restrict__ tq, unsigned short* __restrict__ tk,
    unsigned short* __restrict__ tv)               // [head][L][96]
{
  int idx = blockIdx.x * 256 + threadIdx.x;
  if (idx >= L_TXT * N_HEADS) return;
  int l = idx >> 4, head = idx & 15;
  size_t outOff = ((size_t)head * L_TXT + l) * HD;
  const float* base = qkv + (size_t)l * QKV_N + head * HD;

  float x[96];
#pragma unroll
  for (int d = 0; d < 96; d++) x[d] = base[d];
  rms96(x, gq);
#pragma unroll
  for (int d = 0; d < 96; d++) tq[outOff + d] = f2bf(x[d]);

#pragma unroll
  for (int d = 0; d < 96; d++) x[d] = base[DIM + d];
  rms96(x, gk);
#pragma unroll
  for (int d = 0; d < 96; d++) tk[outOff + d] = f2bf(x[d]);

#pragma unroll
  for (int d = 0; d < 96; d++) tv[outOff + d] = f2bf(base[2 * DIM + d]);
}

// ---------------------------------------------------------------------------
// Flash attention: one wave per 16-row q tile. Keys = nvid vid keys + L txt
// keys, streamed in chunks of 32 through LDS (Q/K staged async). Online
// softmax in fp32.
//   outMode 0: vid — scatter rows back through the window permutation.
//   outMode 1: txt — linear rows.
// ---------------------------------------------------------------------------
__global__ __launch_bounds__(32) void flash_attn_kernel(
    const unsigned short* __restrict__ qbase, long qHeadStride,
    const unsigned short* __restrict__ kvid, const unsigned short* __restrict__ vvid,
    long kvHeadStride, int vidLocal, int nvid,
    const unsigned short* __restrict__ tk, const unsigned short* __restrict__ tv,
    const float* __restrict__ maskbias,
    unsigned short* __restrict__ outp, int outMode)
{
  __shared__ __align__(16) unsigned short ldsQ[16 * HD];
  __shared__ __align__(16) unsigned short ldsK[32 * HD];   // [key][hd]
  __shared__ __align__(16) unsigned short ldsVt[HD * 32];  // [hd][key]
  __shared__ __align__(16) unsigned short ldsP[16 * 32];   // [m][key]

  const int lane = threadIdx.x;
  const int head = blockIdx.y;
  const int ln15 = lane & 15;
  const int halfsel = lane >> 4;
  const int kg = halfsel * 8;

  // ---- Q tile (16 x 96), async staged ----
  const unsigned short* qsrc =
      qbase + (size_t)head * qHeadStride + (size_t)blockIdx.x * 16 * HD;
#pragma unroll
  for (int i = 0; i < 6; i++) {
    int c = lane + i * 32;               // 192 chunks of 8 ushorts
    int r = c / 12, o = (c % 12) * 8;
    async_copy16(&ldsQ[r * HD + o], &qsrc[(size_t)r * HD + o]);
  }
  async_wait0();
  __syncthreads();

  Frag16 qf[3];
#pragma unroll
  for (int ks = 0; ks < 3; ks++)
#pragma unroll
    for (int p = 0; p < 8; p++)
      qf[ks].u[p] = *(const unsigned*)&ldsQ[ln15 * HD + ks * 32 + k0_of(p, kg)];

  v8f zero = {0.f, 0.f, 0.f, 0.f, 0.f, 0.f, 0.f, 0.f};
  v8f o_acc[6];
#pragma unroll
  for (int h = 0; h < 6; h++) o_acc[h] = zero;
  float m8[8], l8[8];
#pragma unroll
  for (int r = 0; r < 8; r++) { m8[r] = -1e30f; l8[r] = 0.f; }

  const size_t kvOff = (size_t)head * kvHeadStride +
                       (vidLocal ? (size_t)(blockIdx.x >> 4) * WIN * HD : (size_t)0);
  const int nKeys = nvid + L_TXT;

  for (int keyBase = 0; keyBase < nKeys; keyBase += 32) {
    const unsigned short *ksrc, *vsrc;
    if (keyBase < nvid) {
      ksrc = kvid + kvOff + (size_t)keyBase * HD;
      vsrc = vvid + kvOff + (size_t)keyBase * HD;
    } else {
      size_t toff = ((size_t)head * L_TXT + (keyBase - nvid)) * HD;
      ksrc = tk + toff;  vsrc = tv + toff;
    }
    // stage K natural [32][96] async; V transposed [96][32] sync
#pragma unroll
    for (int j = 0; j < 12; j++)
      async_copy16(&ldsK[lane * HD + j * 8], &ksrc[(size_t)lane * HD + j * 8]);
    const unsigned* vrow = (const unsigned*)(vsrc + (size_t)lane * HD);
#pragma unroll 8
    for (int j = 0; j < 48; j++) {
      unsigned d = vrow[j];
      ldsVt[(2 * j) * 32 + lane]     = (unsigned short)(d & 0xffffu);
      ldsVt[(2 * j + 1) * 32 + lane] = (unsigned short)(d >> 16);
    }
    async_wait0();
    __syncthreads();

    // ---- scores: q(16x96) . k^T(96x32) via 3 K-steps x 2 key tiles ----
    v8f s0 = zero, s1 = zero;
#pragma unroll
    for (int ks = 0; ks < 3; ks++) {
      Frag16 b0, b1;
#pragma unroll
      for (int p = 0; p < 8; p++) {
        int k0 = k0_of(p, kg);
        b0.u[p] = *(const unsigned*)&ldsK[ln15 * HD + ks * 32 + k0];
        b1.u[p] = *(const unsigned*)&ldsK[(16 + ln15) * HD + ks * 32 + k0];
      }
      s0 = __builtin_amdgcn_wmma_f32_16x16x32_bf16(false, qf[ks].v, false, b0.v,
                                                   (short)0, s0, false, false);
      s1 = __builtin_amdgcn_wmma_f32_16x16x32_bf16(false, qf[ks].v, false, b1.v,
                                                   (short)0, s1, false, false);
    }

    float bias0 = 0.f, bias1 = 0.f;
    int key0 = keyBase + ln15, key1 = keyBase + 16 + ln15;
    if (key0 >= nvid) bias0 = maskbias[key0 - nvid];
    if (key1 >= nvid) bias1 = maskbias[key1 - nvid];

    // ---- online softmax; rows live in 16-lane halves (wave32 C layout) ----
    float alpha8[8];
#pragma unroll
    for (int r = 0; r < 8; r++) {
      float a = s0[r] * ATTN_SCALE + bias0;
      float b = s1[r] * ATTN_SCALE + bias1;
      float mx = fmaxf(a, b);
#pragma unroll
      for (int d = 1; d < 16; d <<= 1) mx = fmaxf(mx, __shfl_xor(mx, d, 32));
      float nm = fmaxf(m8[r], mx);
      float al = __expf(m8[r] - nm);
      float p0 = __expf(a - nm), p1 = __expf(b - nm);
      float sm = p0 + p1;
#pragma unroll
      for (int d = 1; d < 16; d <<= 1) sm += __shfl_xor(sm, d, 32);
      l8[r] = l8[r] * al + sm;
      m8[r] = nm;
      alpha8[r] = al;
      s0[r] = p0;  s1[r] = p1;
    }
#pragma unroll
    for (int h = 0; h < 6; h++)
#pragma unroll
      for (int r = 0; r < 8; r++) o_acc[h][r] *= alpha8[r];

    // ---- P (C layout) -> LDS -> A-fragment layout ----
#pragma unroll
    for (int r = 0; r < 8; r++) {
      int row = r + (halfsel << 3);
      ldsP[row * 32 + ln15]      = f2bf(s0[r]);
      ldsP[row * 32 + 16 + ln15] = f2bf(s1[r]);
    }
    __syncthreads();

    Frag16 pf;
#pragma unroll
    for (int p = 0; p < 8; p++)
      pf.u[p] = *(const unsigned*)&ldsP[ln15 * 32 + k0_of(p, kg)];
#pragma unroll
    for (int h = 0; h < 6; h++) {
      Frag16 vf;
#pragma unroll
      for (int p = 0; p < 8; p++)
        vf.u[p] = *(const unsigned*)&ldsVt[(h * 16 + ln15) * 32 + k0_of(p, kg)];
      o_acc[h] = __builtin_amdgcn_wmma_f32_16x16x32_bf16(false, pf.v, false, vf.v,
                                                         (short)0, o_acc[h], false, false);
    }
    __syncthreads();   // protect ldsK/ldsVt before next chunk staging
  }

  // ---- normalize and write out ----
#pragma unroll
  for (int h = 0; h < 6; h++)
#pragma unroll
    for (int r = 0; r < 8; r++) {
      float val = o_acc[h][r] / l8[r];
      int m = r + (halfsel << 3);
      int col = head * HD + h * 16 + ln15;
      size_t orow;
      if (outMode == 0) {
        int window = blockIdx.x >> 4;
        int rrow = ((blockIdx.x & 15) << 4) + m;
        int it = window / 36, ih = (window / 6) % 6, iw = window % 6;
        int tt = rrow >> 6, hh = (rrow >> 3) & 7, ww = rrow & 7;
        int t = it * 4 + tt, hs = ih * 8 + hh, wsp = iw * 8 + ww;
        orow = (size_t)((t * 48 + hs) * 48 + wsp);
      } else {
        orow = (size_t)(blockIdx.x * 16 + m);
      }
      outp[orow * DIM + col] = f2bf(val);
    }
}

// ---------------------------------------------------------------------------
// Host launch
// ---------------------------------------------------------------------------
extern "C" void kernel_launch(void* const* d_in, const int* in_sizes, int n_in,
                              void* d_out, int out_size, void* d_ws, size_t ws_size,
                              hipStream_t stream) {
  (void)in_sizes; (void)n_in; (void)out_size; (void)ws_size;
  const float* vid        = (const float*)d_in[0];
  const float* txt        = (const float*)d_in[1];
  const unsigned char* txt_mask = (const unsigned char*)d_in[2];
  const float* w_qkv_vid  = (const float*)d_in[3];
  const float* w_qkv_txt  = (const float*)d_in[4];
  const float* w_out_vid  = (const float*)d_in[5];
  const float* b_out_vid  = (const float*)d_in[6];
  const float* w_out_txt  = (const float*)d_in[7];
  const float* b_out_txt  = (const float*)d_in[8];
  const float* gq_vid     = (const float*)d_in[9];
  const float* gq_txt     = (const float*)d_in[10];
  const float* gk_vid     = (const float*)d_in[11];
  const float* gk_txt     = (const float*)d_in[12];

  char* ws = (char*)d_ws;
  size_t cur = 0;
  auto alloc = [&](size_t bytes) -> void* {
    void* p = ws + cur;
    cur += (bytes + 255) & ~(size_t)255;
    return p;
  };

  unsigned short* vid_bf = (unsigned short*)alloc((size_t)S_TOK * DIM * 2);
  unsigned short* txt_bf = (unsigned short*)alloc((size_t)L_TXT * DIM * 2);
  unsigned short* wqv_bf = (unsigned short*)alloc((size_t)DIM * QKV_N * 2);
  unsigned short* wqt_bf = (unsigned short*)alloc((size_t)DIM * QKV_N * 2);
  unsigned short* wov_bf = (unsigned short*)alloc((size_t)DIM * DIM * 2);
  unsigned short* wot_bf = (unsigned short*)alloc((size_t)DIM * DIM * 2);
  float* qkv_vid = (float*)alloc((size_t)S_TOK * QKV_N * 4);
  float* qkv_txt = (float*)alloc((size_t)L_TXT * QKV_N * 4);
  unsigned short* qw = (unsigned short*)alloc((size_t)N_HEADS * S_TOK * HD * 2);
  unsigned short* kw = (unsigned short*)alloc((size_t)N_HEADS * S_TOK * HD * 2);
  unsigned short* vw = (unsigned short*)alloc((size_t)N_HEADS * S_TOK * HD * 2);
  unsigned short* tq = (unsigned short*)alloc((size_t)N_HEADS * L_TXT * HD * 2);
  unsigned short* tk = (unsigned short*)alloc((size_t)N_HEADS * L_TXT * HD * 2);
  unsigned short* tv = (unsigned short*)alloc((size_t)N_HEADS * L_TXT * HD * 2);
  unsigned short* vo   = (unsigned short*)alloc((size_t)S_TOK * DIM * 2);
  unsigned short* to_b = (unsigned short*)alloc((size_t)L_TXT * DIM * 2);
  float* maskbias = (float*)alloc((size_t)L_TXT * 4);

  auto cast = [&](const float* in, unsigned short* out, size_t n) {
    long blocks = (long)((n + 255) / 256);
    if (blocks > 8192) blocks = 8192;
    cast_bf16_kernel<<<dim3((unsigned)blocks), 256, 0, stream>>>(in, out, (long)n);
  };
  cast(vid,       vid_bf, (size_t)S_TOK * DIM);
  cast(txt,       txt_bf, (size_t)L_TXT * DIM);
  cast(w_qkv_vid, wqv_bf, (size_t)DIM * QKV_N);
  cast(w_qkv_txt, wqt_bf, (size_t)DIM * QKV_N);
  cast(w_out_vid, wov_bf, (size_t)DIM * DIM);
  cast(w_out_txt, wot_bf, (size_t)DIM * DIM);
  maskbias_kernel<<<1, 256, 0, stream>>>(txt_mask, maskbias);

  // QKV projections
  gemm_bf16_kernel<<<dim3(QKV_N / 128, S_TOK / 128), 256, 0, stream>>>(
      vid_bf, wqv_bf, qkv_vid, nullptr, S_TOK, QKV_N, DIM);
  gemm_bf16_kernel<<<dim3(QKV_N / 128, L_TXT / 128), 256, 0, stream>>>(
      txt_bf, wqt_bf, qkv_txt, nullptr, L_TXT, QKV_N, DIM);

  // RMS + RoPE + windowing
  post_vid_kernel<<<dim3((S_TOK * N_HEADS) / 256), 256, 0, stream>>>(
      qkv_vid, gq_vid, gk_vid, qw, kw, vw);
  post_txt_kernel<<<dim3((L_TXT * N_HEADS) / 256), 256, 0, stream>>>(
      qkv_txt, gq_txt, gk_txt, tq, tk, tv);

  // Window attention (vid): keys = 256 window keys + 256 txt keys
  flash_attn_kernel<<<dim3(NWIN * 16, N_HEADS), 32, 0, stream>>>(
      qw, (long)NWIN * WIN * HD, kw, vw, (long)NWIN * WIN * HD,
      /*vidLocal=*/1, /*nvid=*/WIN, tk, tv, maskbias, vo, /*outMode=*/0);

  // Txt attention: keys = all 18432 vid keys (any order) + 256 txt keys
  flash_attn_kernel<<<dim3(L_TXT / 16, N_HEADS), 32, 0, stream>>>(
      tq, (long)L_TXT * HD, kw, vw, (long)NWIN * WIN * HD,
      /*vidLocal=*/0, /*nvid=*/S_TOK, tk, tv, maskbias, to_b, /*outMode=*/1);

  // Output projections straight into d_out (vid then txt, fp32)
  float* out_vid = (float*)d_out;
  float* out_txt = out_vid + (size_t)S_TOK * DIM;
  gemm_bf16_kernel<<<dim3(DIM / 128, S_TOK / 128), 256, 0, stream>>>(
      vo, wov_bf, out_vid, b_out_vid, S_TOK, DIM, DIM);
  gemm_bf16_kernel<<<dim3(DIM / 128, L_TXT / 128), 256, 0, stream>>>(
      to_b, wot_bf, out_txt, b_out_txt, L_TXT, DIM, DIM);
}
